// Path_Embedding_69698729279680
// MI455X (gfx1250) — matrix operand order repaired and checked by reading, not compiled, not run on hardware
//
#include <hip/hip_runtime.h>

// ---- types for WMMA operands -------------------------------------------------
typedef _Float16 v8h  __attribute__((ext_vector_type(8)));
typedef _Float16 v16h __attribute__((ext_vector_type(16)));
typedef float    v8f  __attribute__((ext_vector_type(8)));

// ---- geometry ----------------------------------------------------------------
#define NB 8                               // batch rows per block (amortize weights)

// ---- LDS layout --------------------------------------------------------------
// sE : double-buffered 128 gathered embedding rows (p*8+t), f16, stride 136
// sW : B-operand weights, sW[n][c] with c = k*128 + i, f16, stride 264
#define ESTR 136
#define WSTR 264
#define E_BYTES (128 * ESTR * 2)            // 34816 per buffer
#define W_BYTES (128 * WSTR * 2)            // 67584
#define SMEM_BYTES (2 * E_BYTES + W_BYTES)  // 137216 -> 2 blocks / WGP (320KB)

static __device__ __forceinline__ v16h cat16(v8h lo, v8h hi) {
  return __builtin_shufflevector(lo, hi, 0,1,2,3,4,5,6,7,8,9,10,11,12,13,14,15);
}

__global__ void path_emb_conv_max_kernel(
    const int*   __restrict__ path_input,   // (B,16,8) int32
    const int*   __restrict__ path_type,    // (8,) int32 in [0,4)
    const float* __restrict__ user_emb,     // (100000,128)
    const float* __restrict__ item_emb,
    const float* __restrict__ ca_emb,
    const float* __restrict__ ci_emb,
    const float* __restrict__ conv_w,       // (128,128,2)
    const float* __restrict__ conv_b,       // (128,)
    float*       __restrict__ out,          // (B,128)
    int B)
{
  extern __shared__ __align__(16) char smem[];
  _Float16* sE0 = (_Float16*)smem;                       // buffer 0
  _Float16* sE1 = (_Float16*)(smem + E_BYTES);           // buffer 1
  _Float16* sW  = (_Float16*)(smem + 2 * E_BYTES);

  const int tid  = threadIdx.x;
  const int wave = tid >> 5;     // 8 waves (wave32)
  const int lane = tid & 31;

  const int b0 = blockIdx.x * NB;

  // ---- stage weights once per block: sW[n][k*128 + i] = w[n][i][k] ----------
  #pragma unroll 4
  for (int j = 0; j < 128; ++j) {
    int f = tid + j * 256;                  // flat index into (128,128,2)
    int n = f >> 8;
    int i = (f & 255) >> 1;
    int k = f & 1;
    sW[n * WSTR + k * 128 + i] = (_Float16)conv_w[f];
  }

  // ---- gather lambda: 128 rows (r = p*8+t) of batch row b -> f16 LDS buffer -
  auto gather = [&](int b, _Float16* sE) {
    const int rbase = wave * 16;            // 16 rows per wave
    #pragma unroll 4
    for (int rr = 0; rr < 16; ++rr) {
      int r   = rbase + rr;
      int t   = r & 7;
      int idx = path_input[b * 128 + r];
      int pt  = path_type[t];
      const float* tab = (pt == 0) ? user_emb
                       : (pt == 1) ? item_emb
                       : (pt == 2) ? ca_emb : ci_emb;
      const float4 v = *(const float4*)(tab + (size_t)idx * 128 + lane * 4);
      _Float16* dst = sE + r * ESTR + lane * 4;
      dst[0] = (_Float16)v.x;  dst[1] = (_Float16)v.y;
      dst[2] = (_Float16)v.z;  dst[3] = (_Float16)v.w;
    }
  };

  gather(b0, sE0);                          // prime buffer 0
  __syncthreads();

  // ---- wave owns N-tile `wave`: keep its 8 B-fragments in VGPRs forever -----
  const int mrel  = lane & 15;              // A: row-in-tile / B,C: col-in-tile
  const int khalf = lane >> 4;
  const int n     = wave * 16 + mrel;       // this lane's output column
  const float bias = conv_b[n];

  // B fragment layout used here (16-bit 32x16): lane holds N=mrel,
  // K = khalf*16 + {0..15} consecutive -> two ds_load_b128 per fragment.
  v16h bfrag[8];
  {
    const _Float16* wrow = sW + n * WSTR;
    #pragma unroll
    for (int kt = 0; kt < 8; ++kt) {
      int koff = kt * 32 + khalf * 16;
      v8h lo = *(const v8h*)(wrow + koff);
      v8h hi = *(const v8h*)(wrow + koff + 8);
      bfrag[kt] = cat16(lo, hi);
    }
  }

  // ---- pipelined loop over NB batch rows ------------------------------------
  for (int bb = 0; bb < NB; ++bb) {
    _Float16* cur = (bb & 1) ? sE1 : sE0;
    _Float16* nxt = (bb & 1) ? sE0 : sE1;
    int b = b0 + bb;

    // overlap: issue next row's gather before this row's GEMM
    if (bb + 1 < NB && b0 + bb + 1 < B) gather(b0 + bb + 1, nxt);

    // GEMM: M=128 rows ((p,t), K=256 = emb[t] ++ emb[t+1]), this wave's 16 cols.
    // Max-reduce over valid rows entirely in VGPRs.
    float gmax = -3.0e38f;
    #pragma unroll
    for (int mt = 0; mt < 8; ++mt) {
      const int m   = mt * 16 + mrel;             // A row for this lane
      const int mA1 = (m < 127) ? m + 1 : 127;    // row supplying emb[t+1] cols
      v8f acc = {};
      #pragma unroll
      for (int kt = 0; kt < 8; ++kt) {
        // A fragment layout (16-bit 16x32): lane holds K = khalf*16+{0..15}
        // consecutive (matches B) -> two ds_load_b128 per fragment.
        const _Float16* rowp = cur + ((kt < 4) ? m : mA1) * ESTR;
        int coff = (kt & 3) * 32 + khalf * 16;
        v8h lo = *(const v8h*)(rowp + coff);
        v8h hi = *(const v8h*)(rowp + coff + 8);
        v16h afrag = cat16(lo, hi);
        acc = __builtin_amdgcn_wmma_f32_16x16x32_f16(
            /*neg_a=*/false, afrag, /*neg_b=*/false, bfrag[kt],
            /*c_mod=*/(short)0, acc, /*reuse_a=*/false, /*reuse_b=*/false);
      }
      // acc VGPR r holds row m2 = mt*16 + khalf*8 + r  ->  (m2 & 7) == r;
      // exclude t==7 windows (r == 7).
      #pragma unroll
      for (int r = 0; r < 7; ++r) gmax = fmaxf(gmax, acc[r]);
    }

    // combine the two half-wave row groups (same output column n)
    float o = __shfl_xor(gmax, 16, 32);
    gmax = fmaxf(gmax, o);
    if (lane < 16 && b < B) out[(size_t)b * 128 + n] = gmax + bias;

    // one barrier per iteration: next gather's target buffer is rewritten only
    // after the barrier that follows its last GEMM read, and gather(bb+1) completes before
    // GEMM(bb+1) starts.
    __syncthreads();
  }
}

extern "C" void kernel_launch(void* const* d_in, const int* in_sizes, int n_in,
                              void* d_out, int out_size, void* d_ws, size_t ws_size,
                              hipStream_t stream) {
  const int*   path_input = (const int*)d_in[0];
  // d_in[1] = path_num (scalar), d_in[2] = timestamp (scalar) -- unused
  const int*   path_type  = (const int*)d_in[3];
  const float* user_emb   = (const float*)d_in[4];
  const float* item_emb   = (const float*)d_in[5];
  const float* ca_emb     = (const float*)d_in[6];
  const float* ci_emb     = (const float*)d_in[7];
  const float* conv_w     = (const float*)d_in[8];
  const float* conv_b     = (const float*)d_in[9];
  float*       out        = (float*)d_out;

  const int B = in_sizes[0] / 128;          // path_input is (B,16,8)
  const int grid = (B + NB - 1) / NB;

  (void)hipFuncSetAttribute((const void*)path_emb_conv_max_kernel,
                            hipFuncAttributeMaxDynamicSharedMemorySize,
                            SMEM_BYTES);

  path_emb_conv_max_kernel<<<dim3(grid), dim3(256), SMEM_BYTES, stream>>>(
      path_input, path_type, user_emb, item_emb, ca_emb, ci_emb,
      conv_w, conv_b, out, B);
}
